// CrossAttention_23751169147521
// MI455X (gfx1250) — matrix operand order repaired and checked by reading, not compiled, not run on hardware
//
#include <hip/hip_runtime.h>
#include <hip/hip_bf16.h>
#include <stdint.h>

// ---------------------------------------------------------------------------
// CrossAttention for MI455X (gfx1250, wave32, WMMA bf16 16x16x32, f32 accum)
// B=8, C=512, H=W=64 (S=4096), heads=8, dhead=64, Sctx=77, Dctx=768
// ---------------------------------------------------------------------------

typedef __attribute__((ext_vector_type(16))) __bf16 v16bf;
typedef __attribute__((ext_vector_type(8)))  float  v8f;

#define BATCH   8
#define CCH     512
#define SEQ     4096
#define HEADS   8
#define DHEAD   64
#define INNER   512
#define SCTX    77
#define SKPAD   80     // Sctx padded to 5 n-tiles of 16 for QK^T
#define SKPAD2  96     // Sctx padded to 3 k-steps of 32 for attn@V
#define DCTX    768
#define SCALEF  0.125f // 64^-0.5

__device__ __forceinline__ __bf16 f2bf(float x) {
  uint32_t u = __float_as_uint(x);
  u = (u + 0x7FFFu + ((u >> 16) & 1u)) >> 16;   // round-to-nearest-even
  uint16_t h = (uint16_t)u;
  __bf16 b;
  __builtin_memcpy(&b, &h, 2);
  return b;
}

union FragU { v16bf v; uint4 q[2]; };

// A-fragment (16x32 bf16) from row-major [rows, ld]: lane l -> row row0+(l&15),
// elements 0..7 = K[kb..kb+7], 8..15 = K[kb+16..kb+23], kb = k0 + (l>=16)*8.
__device__ __forceinline__ v16bf load_frag_a(const __bf16* base, int ld,
                                             int row0, int k0, int lane) {
  const int m  = row0 + (lane & 15);
  const int kb = k0 + ((lane >> 4) << 3);
  const __bf16* p = base + (size_t)m * ld + kb;
  FragU u;
  u.q[0] = *(const uint4*)(p);        // K kb..kb+7
  u.q[1] = *(const uint4*)(p + 16);   // K kb+16..kb+23
  return u.v;
}

// B-fragment (32x16 bf16) loaded from B^T stored row-major [N, ld]:
// lane l -> column n0+(l&15), elements 0..7 = K[kb..kb+7], 8..15 =
// K[kb+16..kb+23], kb = k0 + (l>=16)*8.
__device__ __forceinline__ v16bf load_frag_b(const __bf16* baseT, int ld,
                                             int n0, int k0, int lane) {
  const int n  = n0 + (lane & 15);
  const int kb = k0 + ((lane >> 4) << 3);
  const __bf16* p = baseT + (size_t)n * ld + kb;
  FragU u;
  u.q[0] = *(const uint4*)(p);
  u.q[1] = *(const uint4*)(p + 16);
  return u.v;
}

__device__ __forceinline__ v8f wmma_bf16(v16bf a, v16bf b, v8f c) {
  return __builtin_amdgcn_wmma_f32_16x16x32_bf16(false, a, false, b,
                                                 (short)0, c, false, false);
}

__device__ __forceinline__ v8f vzero8() {
  v8f z;
#pragma unroll
  for (int i = 0; i < 8; ++i) z[i] = 0.0f;
  return z;
}

// ---------------------------------------------------------------------------
// K0: transpose Wq (512x512) -> WqT bf16 and Wo -> WoT bf16.  grid (16,16,2)
// ---------------------------------------------------------------------------
__global__ void wtrans_kernel(const float* __restrict__ Wq,
                              const float* __restrict__ Wo,
                              __bf16* __restrict__ WqT,
                              __bf16* __restrict__ WoT) {
  __shared__ float tile[32][33];
  const float* src = (blockIdx.z == 0) ? Wq : Wo;
  __bf16*      dst = (blockIdx.z == 0) ? WqT : WoT;
  const int k0 = blockIdx.x * 32;   // src row (input channel)
  const int n0 = blockIdx.y * 32;   // src col (output channel)
  const int tx = threadIdx.x, ty = threadIdx.y;
#pragma unroll
  for (int i = 0; i < 32; i += 8)
    tile[ty + i][tx] = src[(size_t)(k0 + ty + i) * CCH + n0 + tx];
  __syncthreads();
#pragma unroll
  for (int i = 0; i < 32; i += 8)
    dst[(size_t)(n0 + ty + i) * CCH + k0 + tx] = f2bf(tile[tx][ty + i]);
}

// ---------------------------------------------------------------------------
// K1: xT[b,s,c] = bf16(x[b,c,s]).  grid (128,16,8), block (32,8)
// ---------------------------------------------------------------------------
__global__ void xtrans_kernel(const float* __restrict__ x,
                              __bf16* __restrict__ xT) {
  __shared__ float tile[32][33];
  const int b  = blockIdx.z;
  const int s0 = blockIdx.x * 32;
  const int c0 = blockIdx.y * 32;
  const float* xb  = x  + (size_t)b * CCH * SEQ;
  __bf16*      xTb = xT + (size_t)b * SEQ * CCH;
  const int tx = threadIdx.x, ty = threadIdx.y;
#pragma unroll
  for (int i = 0; i < 32; i += 8)
    tile[ty + i][tx] = xb[(size_t)(c0 + ty + i) * SEQ + s0 + tx];
  __syncthreads();
#pragma unroll
  for (int i = 0; i < 32; i += 8)
    xTb[(size_t)(s0 + ty + i) * CCH + c0 + tx] = f2bf(tile[tx][ty + i]);
}

// ---------------------------------------------------------------------------
// K2: K/V projection (tiny: ~1 GFLOP, scalar f32).
//   Kmat[b, j(0..79), i]     = ctx[b,j,:] . Wk[:,i]   (0 for j>=77)
//   Vt  [b, h, d, j(0..95)]  = ctx[b,j,:] . Wv[:,h*64+d]  (0 for j>=77)
// grid (2, 96, 8), block 256
// ---------------------------------------------------------------------------
__global__ void kv_kernel(const float* __restrict__ ctx,
                          const float* __restrict__ Wk,
                          const float* __restrict__ Wv,
                          __bf16* __restrict__ Kmat,
                          __bf16* __restrict__ Vt) {
  const int b = blockIdx.z;
  const int j = blockIdx.y;                            // 0..95
  const int i = blockIdx.x * blockDim.x + threadIdx.x; // 0..511
  float kv = 0.0f, vv = 0.0f;
  if (j < SCTX) {
    const float* crow = ctx + ((size_t)b * SCTX + j) * DCTX;
    for (int t = 0; t < DCTX; ++t) {
      const float c = crow[t];
      kv = fmaf(c, Wk[(size_t)t * INNER + i], kv);
      vv = fmaf(c, Wv[(size_t)t * INNER + i], vv);
    }
  }
  if (j < SKPAD)
    Kmat[((size_t)b * SKPAD + j) * INNER + i] = f2bf(kv);
  const int h = i >> 6, d = i & 63;
  Vt[(((size_t)b * HEADS + h) * DHEAD + d) * SKPAD2 + j] = f2bf(vv);
}

// ---------------------------------------------------------------------------
// K3: Q = (xT @ Wq) * scale, bf16 out.
// Per wave: 32x64 tile (2 A-row frags share each B frag -> 8 WMMAs/k-step).
// Single live fragment set (acc 64 + A 16 + B 8 VGPRs) -> no spills; the
// fully-unrolled k-loop lets the compiler hoist loads across WMMAs itself.
// __launch_bounds__(256,1): prefer registers over occupancy (no scratch).
// grid (16, 8, 8), block 256 (8 waves)
// ---------------------------------------------------------------------------
__global__ void __launch_bounds__(256, 1)
qproj_kernel(const __bf16* __restrict__ xT,
             const __bf16* __restrict__ WqT,
             __bf16* __restrict__ Q) {
  const int lane = threadIdx.x & 31;
  const int w    = threadIdx.x >> 5;
  const int b    = blockIdx.z;
  const int m0   = (blockIdx.x * 8 + w) * 32;
  const int n0   = blockIdx.y * 64;
  const __bf16* A = xT + (size_t)b * SEQ * CCH;

  v8f acc[2][4];
#pragma unroll
  for (int i = 0; i < 2; ++i)
#pragma unroll
    for (int t = 0; t < 4; ++t) acc[i][t] = vzero8();

  for (int k = 0; k < CCH; k += 32) {
    if (k + 64 < CCH)
      __builtin_prefetch(A + (size_t)(m0 + (lane & 15)) * CCH + k + 64, 0, 1);
    const v16bf a0 = load_frag_a(A, CCH, m0,      k, lane);
    const v16bf a1 = load_frag_a(A, CCH, m0 + 16, k, lane);
#pragma unroll
    for (int t = 0; t < 4; ++t) {
      const v16bf bf = load_frag_b(WqT, CCH, n0 + t * 16, k, lane);
      acc[0][t] = wmma_bf16(a0, bf, acc[0][t]);
      acc[1][t] = wmma_bf16(a1, bf, acc[1][t]);
    }
  }

  __bf16* Qb = Q + (size_t)b * SEQ * INNER;
  const int rowbase = (lane >> 4) << 3;
#pragma unroll
  for (int i = 0; i < 2; ++i) {
    const int mbase = m0 + i * 16 + rowbase;
#pragma unroll
    for (int t = 0; t < 4; ++t) {
      const int col = n0 + t * 16 + (lane & 15);
#pragma unroll
      for (int r = 0; r < 8; ++r)
        Qb[(size_t)(mbase + r) * INNER + col] = f2bf(acc[i][t][r] * SCALEF);
    }
  }
}

// ---------------------------------------------------------------------------
// K4: fused attention. Per wave: one (b, h, 16-row) tile.
//   sim(16x80) = Qh(16x64) @ Kh^T        -> 10 WMMAs
//   softmax over 77 cols (in-register, shfl_xor butterflies within 16-lane halves)
//   O(16x64)  = P(16x96) @ Vh(96x64)     -> 12 WMMAs (P staged in LDS)
// grid 2048, block 256 (8 waves); LDS: 8 * 16*96 bf16 = 24 KB
// ---------------------------------------------------------------------------
__global__ void attn_kernel(const __bf16* __restrict__ Q,
                            const __bf16* __restrict__ Kmat,
                            const __bf16* __restrict__ Vt,
                            __bf16* __restrict__ O) {
  __shared__ __bf16 P[8][16 * SKPAD2];
  const int lane = threadIdx.x & 31;
  const int w    = threadIdx.x >> 5;
  const int id   = blockIdx.x * 8 + w;     // 0..16383
  const int b    = id >> 11;
  const int rem  = id & 2047;
  const int h    = rem >> 8;
  const int m0   = (rem & 255) << 4;

  const __bf16* Qh = Q    + (size_t)b * SEQ * INNER + h * DHEAD;
  const __bf16* Kh = Kmat + (size_t)b * SKPAD * INNER + h * DHEAD;
  const __bf16* Vh = Vt   + ((size_t)(b * HEADS + h)) * DHEAD * SKPAD2;

  const v16bf qa0 = load_frag_a(Qh, INNER, m0, 0, lane);
  const v16bf qa1 = load_frag_a(Qh, INNER, m0, 32, lane);

  v8f s[5];
#pragma unroll
  for (int j = 0; j < 5; ++j) {
    v8f acc = vzero8();
    const v16bf b0 = load_frag_b(Kh, INNER, j * 16, 0, lane);
    acc = wmma_bf16(qa0, b0, acc);
    const v16bf b1 = load_frag_b(Kh, INNER, j * 16, 32, lane);
    acc = wmma_bf16(qa1, b1, acc);
    s[j] = acc;
  }

  // ---- softmax over columns (rows live split across 16-lane halves) ----
  const int nc = lane & 15;
  float mx[8], sum[8];
#pragma unroll
  for (int r = 0; r < 8; ++r) { mx[r] = -1e30f; sum[r] = 0.0f; }
#pragma unroll
  for (int j = 0; j < 5; ++j) {
    const bool masked = (j * 16 + nc) >= SCTX;
#pragma unroll
    for (int r = 0; r < 8; ++r) {
      if (masked) s[j][r] = -1e30f;
      mx[r] = fmaxf(mx[r], s[j][r]);
    }
  }
#pragma unroll
  for (int off = 1; off <= 8; off <<= 1)
#pragma unroll
    for (int r = 0; r < 8; ++r)
      mx[r] = fmaxf(mx[r], __shfl_xor(mx[r], off, 32));
#pragma unroll
  for (int j = 0; j < 5; ++j)
#pragma unroll
    for (int r = 0; r < 8; ++r) {
      const float p = __expf(s[j][r] - mx[r]);   // masked -> exp(-huge) = 0
      s[j][r] = p;
      sum[r] += p;
    }
#pragma unroll
  for (int off = 1; off <= 8; off <<= 1)
#pragma unroll
    for (int r = 0; r < 8; ++r)
      sum[r] += __shfl_xor(sum[r], off, 32);

  // ---- stage P (16x96, bf16, zero-padded cols 80..95) into LDS ----
  __bf16* Pw = P[w];
  const int rowbase = (lane >> 4) << 3;
#pragma unroll
  for (int j = 0; j < 5; ++j) {
    const int col = j * 16 + nc;
#pragma unroll
    for (int r = 0; r < 8; ++r)
      Pw[(rowbase + r) * SKPAD2 + col] = f2bf(s[j][r] / sum[r]);
  }
  {
    const int col = 80 + nc;
#pragma unroll
    for (int r = 0; r < 8; ++r)
      Pw[(rowbase + r) * SKPAD2 + col] = f2bf(0.0f);
  }
  __syncthreads();

  // ---- O = P @ V  (k = 96 in 3 steps of 32) ----
  v8f o[4];
#pragma unroll
  for (int t = 0; t < 4; ++t) o[t] = vzero8();
#pragma unroll
  for (int kk = 0; kk < 3; ++kk) {
    const int k = kk * 32;
    const v16bf pa = load_frag_a(Pw, SKPAD2, 0, k, lane);
#pragma unroll
    for (int t = 0; t < 4; ++t) {
      const v16bf vb = load_frag_b(Vh, SKPAD2, t * 16, k, lane);
      o[t] = wmma_bf16(pa, vb, o[t]);
    }
  }

  __bf16* Ob = O + (size_t)b * SEQ * INNER + h * DHEAD;
  const int mbase = m0 + rowbase;
#pragma unroll
  for (int t = 0; t < 4; ++t) {
    const int d = t * 16 + nc;
#pragma unroll
    for (int r = 0; r < 8; ++r)
      Ob[(size_t)(mbase + r) * INNER + d] = f2bf(o[t][r]);
  }
}

// ---------------------------------------------------------------------------
// K5: out[b,c,s] = (O @ Wo)[s,c] + bo[c].  Per wave: 32x64 tile.
// C/D layout gives 8 consecutive s per lane -> two float4 stores per frag.
// grid (16, 8, 8), block 256
// ---------------------------------------------------------------------------
__global__ void __launch_bounds__(256, 1)
oproj_kernel(const __bf16* __restrict__ O,
             const __bf16* __restrict__ WoT,
             const float* __restrict__ bo,
             float* __restrict__ out) {
  const int lane = threadIdx.x & 31;
  const int w    = threadIdx.x >> 5;
  const int b    = blockIdx.z;
  const int m0   = (blockIdx.x * 8 + w) * 32;
  const int n0   = blockIdx.y * 64;
  const __bf16* A = O + (size_t)b * SEQ * INNER;

  v8f acc[2][4];
#pragma unroll
  for (int i = 0; i < 2; ++i)
#pragma unroll
    for (int t = 0; t < 4; ++t) acc[i][t] = vzero8();

  for (int k = 0; k < INNER; k += 32) {
    if (k + 64 < INNER)
      __builtin_prefetch(A + (size_t)(m0 + (lane & 15)) * INNER + k + 64, 0, 1);
    const v16bf a0 = load_frag_a(A, INNER, m0,      k, lane);
    const v16bf a1 = load_frag_a(A, INNER, m0 + 16, k, lane);
#pragma unroll
    for (int t = 0; t < 4; ++t) {
      const v16bf bf = load_frag_b(WoT, INNER, n0 + t * 16, k, lane);
      acc[0][t] = wmma_bf16(a0, bf, acc[0][t]);
      acc[1][t] = wmma_bf16(a1, bf, acc[1][t]);
    }
  }

  float* ob = out + (size_t)b * CCH * SEQ;
  const int rowbase = (lane >> 4) << 3;
#pragma unroll
  for (int i = 0; i < 2; ++i) {
    const int mbase = m0 + i * 16 + rowbase;
#pragma unroll
    for (int t = 0; t < 4; ++t) {
      const int c = n0 + t * 16 + (lane & 15);
      const float bias = bo[c];
      float4 lo = make_float4(acc[i][t][0] + bias, acc[i][t][1] + bias,
                              acc[i][t][2] + bias, acc[i][t][3] + bias);
      float4 hi = make_float4(acc[i][t][4] + bias, acc[i][t][5] + bias,
                              acc[i][t][6] + bias, acc[i][t][7] + bias);
      *(float4*)(ob + (size_t)c * SEQ + mbase)     = lo;
      *(float4*)(ob + (size_t)c * SEQ + mbase + 4) = hi;
    }
  }
}

// ---------------------------------------------------------------------------
// Launch
// ---------------------------------------------------------------------------
extern "C" void kernel_launch(void* const* d_in, const int* in_sizes, int n_in,
                              void* d_out, int out_size, void* d_ws, size_t ws_size,
                              hipStream_t stream) {
  const float* x    = (const float*)d_in[0];
  const float* ctx  = (const float*)d_in[1];
  const float* Wq   = (const float*)d_in[2];
  const float* Wk   = (const float*)d_in[3];
  const float* Wv   = (const float*)d_in[4];
  const float* Wo   = (const float*)d_in[5];
  const float* bo   = (const float*)d_in[6];
  float* out = (float*)d_out;

  char* ws = (char*)d_ws;
  size_t off = 0;
  auto carve = [&](size_t bytes) -> char* {
    char* p = ws + off;
    off += (bytes + 255) & ~(size_t)255;
    return p;
  };
  __bf16* xT   = (__bf16*)carve((size_t)BATCH * SEQ * CCH * 2);          // 32 MB
  __bf16* Qm   = (__bf16*)carve((size_t)BATCH * SEQ * INNER * 2);        // 32 MB
  __bf16* Om   = (__bf16*)carve((size_t)BATCH * SEQ * INNER * 2);        // 32 MB
  __bf16* WqT  = (__bf16*)carve((size_t)CCH * INNER * 2);
  __bf16* WoT  = (__bf16*)carve((size_t)INNER * CCH * 2);
  __bf16* Kmat = (__bf16*)carve((size_t)BATCH * SKPAD * INNER * 2);
  __bf16* Vtm  = (__bf16*)carve((size_t)BATCH * HEADS * DHEAD * SKPAD2 * 2);
  (void)in_sizes; (void)n_in; (void)out_size; (void)ws_size;

  wtrans_kernel<<<dim3(16, 16, 2), dim3(32, 8), 0, stream>>>(Wq, Wo, WqT, WoT);
  xtrans_kernel<<<dim3(SEQ / 32, CCH / 32, BATCH), dim3(32, 8), 0, stream>>>(x, xT);
  kv_kernel<<<dim3(2, SKPAD2, BATCH), dim3(256), 0, stream>>>(ctx, Wk, Wv, Kmat, Vtm);
  qproj_kernel<<<dim3(16, 8, BATCH), dim3(256), 0, stream>>>(xT, WqT, Qm);
  attn_kernel<<<dim3(2048), dim3(256), 0, stream>>>(Qm, Kmat, Vtm, Om);
  oproj_kernel<<<dim3(16, 8, BATCH), dim3(256), 0, stream>>>(Om, WoT, bo, out);
}
